// EnhancedBoundaryAttnPool_17111149707978
// MI455X (gfx1250) — matrix operand order repaired and checked by reading, not compiled, not run on hardware
//
#include <hip/hip_runtime.h>
#include <cstdint>
#include <cstddef>

// ---------------- problem constants ----------------
#define Bc   16
#define Tc   2048
#define Kc   1024
#define Hc   1024
#define NHc  8
#define Dc   128            // head dim
#define NEG_BIG (-1.0e30f)

typedef __bf16  bf16_t;
typedef __attribute__((ext_vector_type(16))) __bf16 v16bf;
typedef __attribute__((ext_vector_type(8)))  float  v8f;

typedef unsigned short ushort_t;

// ---------------- helpers ----------------
__device__ __forceinline__ unsigned short f2bf(float f) {
    unsigned int u = __builtin_bit_cast(unsigned int, f);
    unsigned int lsb = (u >> 16) & 1u;
    u += 0x7fffu + lsb;                       // round-to-nearest-even
    return (unsigned short)(u >> 16);
}
__device__ __forceinline__ float bf2f(unsigned short h) {
    unsigned int u = ((unsigned int)h) << 16;
    return __builtin_bit_cast(float, u);
}

// CDNA5 async global->LDS copy (16B per lane), tracked by ASYNCcnt.
// Flat pointers to LDS carry the group-relative offset in addr[31:0].
__device__ __forceinline__ void async_copy_b128(void* lds_dst, const void* gsrc) {
    unsigned loff = (unsigned)(size_t)lds_dst;
    asm volatile("global_load_async_to_lds_b128 %0, %1, off"
                 :: "v"(loff), "v"(gsrc) : "memory");
}
__device__ __forceinline__ void wait_async_le4() {
    asm volatile("s_wait_asynccnt 0x4" ::: "memory");
}
__device__ __forceinline__ void wait_async_0() {
    asm volatile("s_wait_asynccnt 0x0" ::: "memory");
}

// ---------------- conversion kernels ----------------
__global__ __launch_bounds__(256) void f32_to_bf16_kernel(const float* __restrict__ x,
                                                          ushort_t* __restrict__ y,
                                                          size_t n4) {
    size_t i = (size_t)blockIdx.x * 256 + threadIdx.x;
    if (i >= n4) return;
    float4 v = ((const float4*)x)[i];
    ushort_t* o = y + i * 4;
    o[0] = f2bf(v.x); o[1] = f2bf(v.y); o[2] = f2bf(v.z); o[3] = f2bf(v.w);
}

// W: (Nout x Kd) f32 row-major  ->  Wt: (Kd x Nout) bf16 row-major
__global__ __launch_bounds__(256) void transpose_w_kernel(const float* __restrict__ W,
                                                          ushort_t* __restrict__ Wt,
                                                          int Nout, int Kd) {
    int idx = blockIdx.x * 256 + threadIdx.x;
    if (idx >= Nout * Kd) return;
    int k = idx / Nout;
    int n = idx - k * Nout;
    Wt[idx] = f2bf(W[(size_t)n * Kd + k]);
}

// ---------------- segment-mean pooling ----------------
__global__ __launch_bounds__(256) void pool_kernel(const float* __restrict__ proj,
                                                   const long long* __restrict__ bnd,
                                                   const float* __restrict__ smask,
                                                   ushort_t* __restrict__ out_bf) {
    int slot = blockIdx.x;                 // b*Kc + k
    int b = slot / Kc;
    long long s = bnd[(size_t)slot * 2 + 0];
    long long e = bnd[(size_t)slot * 2 + 1];
    bool valid = smask[slot] > 0.5f;
    int cnt = (int)(e - s);
    if (cnt < 0) cnt = 0;
    float denom = fmaxf(valid ? (float)cnt : 0.0f, 1.0f);
    int col = threadIdx.x * 4;
    float4 acc = {0.f, 0.f, 0.f, 0.f};
    if (valid) {
        for (int t = 0; t < cnt; ++t) {
            const float4 v = *(const float4*)(proj + ((size_t)b * Tc + (size_t)(s + t)) * Hc + col);
            acc.x += v.x; acc.y += v.y; acc.z += v.z; acc.w += v.w;
        }
    }
    float inv = 1.0f / denom;
    ushort_t* o = out_bf + (size_t)slot * Hc + col;
    o[0] = f2bf(acc.x * inv); o[1] = f2bf(acc.y * inv);
    o[2] = f2bf(acc.z * inv); o[3] = f2bf(acc.w * inv);
}

// ---------------- WMMA GEMM: out = A(bf16, MxK) * Wt(bf16, KxN) + bias ----------------
// block tile 128x128, k-step 32, 8 waves each owning a 64x32 sub-tile.
// Double-buffered LDS fed by global_load_async_to_lds_b128 (ASYNCcnt pipelined).
__global__ __launch_bounds__(256) void gemm_bf16_kernel(const ushort_t* __restrict__ A, int lda,
                                                        const ushort_t* __restrict__ Wt, int ldw,
                                                        const float* __restrict__ bias,
                                                        float* __restrict__ outF,
                                                        ushort_t* __restrict__ outB, int ldo,
                                                        int M, int N, int Kd) {
    __shared__ ushort_t Asm[2][128][48];   // 32 k-cols, pad to 48 (96B rows, 32B-aligned chunks)
    __shared__ ushort_t Bsm[2][32][144];   // 128 n-cols, pad to 144 (288B rows)

    const int tid  = threadIdx.x;
    const int wave = tid >> 5;
    const int lane = tid & 31;
    const int bm = blockIdx.x * 128;
    const int bn = blockIdx.y * 128;
    const int wm = (wave >> 2) * 64;    // 0 / 64
    const int wn = (wave & 3)  * 32;    // 0..96

    v8f acc[4][2];
    for (int i = 0; i < 4; ++i)
        for (int j = 0; j < 2; ++j)
            acc[i][j] = (v8f){0.f,0.f,0.f,0.f,0.f,0.f,0.f,0.f};

    const int ar = tid >> 1;            // A stage: 2 threads per row
    const int ac = (tid & 1) * 16;
    const int br = tid >> 3;            // B stage: 8 threads per row
    const int bc = (tid & 7) * 16;

    // async-stage one 128x32 A slab + 32x128 B slab into LDS buffer `buf`
    auto stage = [&](int buf, int k0) {
        const ushort_t* asrc = A  + (size_t)(bm + ar) * lda + k0 + ac;
        const ushort_t* bsrc = Wt + (size_t)(k0 + br) * ldw + bn + bc;
        async_copy_b128(&Asm[buf][ar][ac],     asrc);
        async_copy_b128(&Asm[buf][ar][ac + 8], asrc + 8);
        async_copy_b128(&Bsm[buf][br][bc],     bsrc);
        async_copy_b128(&Bsm[buf][br][bc + 8], bsrc + 8);
    };

    const int nk = Kd >> 5;
    stage(0, 0);
    for (int ki = 0; ki < nk; ++ki) {
        const int buf = ki & 1;
        if (ki + 1 < nk) {
            stage(buf ^ 1, (ki + 1) * 32);          // prefetch next slab (async)
            wait_async_le4();                       // 4 newest outstanding => slab ki landed
            if (ki + 2 < nk)                        // warm L2 two slabs ahead
                __builtin_prefetch(A + (size_t)(bm + ar) * lda + (ki + 2) * 32 + ac, 0, 1);
        } else {
            wait_async_0();
        }
        __syncthreads();

        // fragments: A lane = row (m%16), K-half by lane/16 ; B lane = col (n%16), K-half by lane/16
        const int arow  = lane & 15;
        const int ahalf = (lane >> 4) * 16;
        v16bf afrag[4], bfrag[2];
        for (int mt = 0; mt < 4; ++mt)
            afrag[mt] = *(const v16bf*)&Asm[buf][wm + mt * 16 + arow][ahalf];
        for (int nt = 0; nt < 2; ++nt)
            for (int s = 0; s < 16; ++s)
                bfrag[nt][s] = __builtin_bit_cast(__bf16, Bsm[buf][ahalf + s][wn + nt * 16 + arow]);

        for (int mt = 0; mt < 4; ++mt)
            for (int nt = 0; nt < 2; ++nt)
                acc[mt][nt] = __builtin_amdgcn_wmma_f32_16x16x32_bf16(
                    false, afrag[mt], false, bfrag[nt],
                    (short)0, acc[mt][nt], false, false);
        __syncthreads();   // protects the buffer the next iteration overwrites
    }

    // epilogue: D element (m,n): n = lane%16, m = vgpr + 8*(lane>=16)
    const int dn    = lane & 15;
    const int mbase = (lane >> 4) * 8;
    for (int mt = 0; mt < 4; ++mt) {
        for (int nt = 0; nt < 2; ++nt) {
            const int gn = bn + wn + nt * 16 + dn;
            const float bv = bias ? bias[gn] : 0.0f;
            for (int v = 0; v < 8; ++v) {
                const int gm = bm + wm + mt * 16 + mbase + v;
                const float val = acc[mt][nt][v] + bv;
                const size_t idx = (size_t)gm * ldo + gn;
                if (outF) outF[idx] = val;
                if (outB) outB[idx] = f2bf(val);
            }
        }
    }
}

// ---------------- windowed cross-attention ----------------
// block = (b, slot), 8 waves = 8 heads; window width <= 32 -> lane j owns key j
__global__ __launch_bounds__(256) void cross_attn_kernel(const ushort_t* __restrict__ qproj, // (B,K,H)
                                                         const ushort_t* __restrict__ kv,    // (B,T,2H)
                                                         const long long* __restrict__ bnd,
                                                         const float* __restrict__ smask,
                                                         ushort_t* __restrict__ o_out) {     // (B,K,H)
    const int slot = blockIdx.x;
    const int b    = slot / Kc;
    const int head = threadIdx.x >> 5;
    const int lane = threadIdx.x & 31;

    long long s = bnd[(size_t)slot * 2 + 0];
    long long e = bnd[(size_t)slot * 2 + 1];
    int width = (int)(e - s);
    if (width > 32) width = 32;
    if (width < 0)  width = 0;
    const bool valid = smask[slot] > 0.5f;

    const ushort4 qd = *(const ushort4*)(qproj + (size_t)slot * Hc + head * Dc + lane * 4);
    const float4 qv = {bf2f(qd.x), bf2f(qd.y), bf2f(qd.z), bf2f(qd.w)};

    float myscore = NEG_BIG;
    for (int j = 0; j < width; ++j) {
        const ushort4 kd = *(const ushort4*)(kv + ((size_t)b * Tc + (size_t)(s + j)) * (2 * Hc)
                                                + head * Dc + lane * 4);
        float p = qv.x * bf2f(kd.x) + qv.y * bf2f(kd.y) + qv.z * bf2f(kd.z) + qv.w * bf2f(kd.w);
        for (int off = 16; off; off >>= 1) p += __shfl_xor(p, off, 32);
        if (lane == j) myscore = p * 0.08838834764831845f;   // 1/sqrt(128)
    }
    if (!valid || lane >= width) myscore = NEG_BIG;

    float m = myscore;
    for (int off = 16; off; off >>= 1) m = fmaxf(m, __shfl_xor(m, off, 32));
    float p = (valid && lane < width) ? __expf(myscore - m) : 0.0f;
    float sum = p;
    for (int off = 16; off; off >>= 1) sum += __shfl_xor(sum, off, 32);
    const float inv = (sum > 0.0f) ? 1.0f / sum : 0.0f;

    float4 o = {0.f, 0.f, 0.f, 0.f};
    for (int j = 0; j < width; ++j) {
        const float pj = __shfl(p, j, 32) * inv;
        const ushort4 vd = *(const ushort4*)(kv + ((size_t)b * Tc + (size_t)(s + j)) * (2 * Hc)
                                                + Hc + head * Dc + lane * 4);
        o.x += pj * bf2f(vd.x); o.y += pj * bf2f(vd.y);
        o.z += pj * bf2f(vd.z); o.w += pj * bf2f(vd.w);
    }
    ushort_t* op = o_out + (size_t)slot * Hc + head * Dc + lane * 4;
    op[0] = f2bf(o.x); op[1] = f2bf(o.y); op[2] = f2bf(o.z); op[3] = f2bf(o.w);
}

// ---------------- fused residual + LayerNorm ----------------
__global__ __launch_bounds__(256) void add_ln_kernel(const float* __restrict__ x,
                                                     const float* __restrict__ r,
                                                     const float* __restrict__ g,
                                                     const float* __restrict__ bta,
                                                     float* __restrict__ outF,
                                                     ushort_t* __restrict__ outB) {
    __shared__ float red[8];
    __shared__ float stat[2];
    const int row = blockIdx.x;
    const int tid = threadIdx.x;
    const int col = tid * 4;

    const float4 xv = *(const float4*)(x + (size_t)row * Hc + col);
    const float4 rv = *(const float4*)(r + (size_t)row * Hc + col);
    float4 t = {xv.x + rv.x, xv.y + rv.y, xv.z + rv.z, xv.w + rv.w};

    float s = t.x + t.y + t.z + t.w;
    for (int off = 16; off; off >>= 1) s += __shfl_xor(s, off, 32);
    if ((tid & 31) == 0) red[tid >> 5] = s;
    __syncthreads();
    if (tid == 0) {
        float tot = 0.f;
        for (int i = 0; i < 8; ++i) tot += red[i];
        stat[0] = tot * (1.0f / Hc);
    }
    __syncthreads();
    const float mean = stat[0];

    const float d0 = t.x - mean, d1 = t.y - mean, d2 = t.z - mean, d3 = t.w - mean;
    float sq = d0 * d0 + d1 * d1 + d2 * d2 + d3 * d3;
    for (int off = 16; off; off >>= 1) sq += __shfl_xor(sq, off, 32);
    __syncthreads();
    if ((tid & 31) == 0) red[tid >> 5] = sq;
    __syncthreads();
    if (tid == 0) {
        float tot = 0.f;
        for (int i = 0; i < 8; ++i) tot += red[i];
        stat[1] = rsqrtf(tot * (1.0f / Hc) + 1e-5f);
    }
    __syncthreads();
    const float inv = stat[1];

    const float4 gv = *(const float4*)(g + col);
    const float4 bv = *(const float4*)(bta + col);
    float4 o = {d0 * inv * gv.x + bv.x, d1 * inv * gv.y + bv.y,
                d2 * inv * gv.z + bv.z, d3 * inv * gv.w + bv.w};
    if (outF) *(float4*)(outF + (size_t)row * Hc + col) = o;
    if (outB) {
        ushort_t* ob = outB + (size_t)row * Hc + col;
        ob[0] = f2bf(o.x); ob[1] = f2bf(o.y); ob[2] = f2bf(o.z); ob[3] = f2bf(o.w);
    }
}

// ---------------- flash causal self-attention over slots ----------------
// 1 wave per block; block = (b, head, 16-row q-tile); K=1024 keys in 32-key chunks
__global__ __launch_bounds__(32) void self_attn_kernel(const ushort_t* __restrict__ qkv, // (B,K,3H)
                                                       const float* __restrict__ smask, // (B,K)
                                                       ushort_t* __restrict__ o_out) {  // (B,K,H)
    const int bid = blockIdx.x;
    const int qt = bid & 63;
    const int h  = (bid >> 6) & 7;
    const int b  = bid >> 9;
    const int m0 = qt * 16;
    const int lane = threadIdx.x;

    __shared__ ushort_t Qs[16][Dc + 16];   // stride 144 (288B)
    __shared__ ushort_t Kts[Dc][48];       // transposed K chunk: [d][key], stride 48 (96B)
    __shared__ ushort_t Vs[32][Dc + 16];   // [key][d]
    __shared__ ushort_t Ps[16][48];        // probabilities (bf16) for PV WMMA
    __shared__ float    Ss[16][34];
    __shared__ float    mrow[16], lrow[16], arow[16], smrow[32];

    const size_t base = (size_t)b * Kc * (3 * Hc);

    // stage Q tile (16 x 128) via async-to-LDS
    {
        const int r  = lane >> 1;
        const int c0 = (lane & 1) * 64;
        const ushort_t* src = qkv + base + (size_t)(m0 + r) * (3 * Hc) + h * Dc + c0;
        for (int i = 0; i < 8; ++i)
            async_copy_b128(&Qs[r][c0 + i * 8], src + i * 8);
    }
    if (lane < 16) { mrow[lane] = NEG_BIG; lrow[lane] = 0.0f; }
    wait_async_0();
    __syncthreads();

    v16bf qfrag[4];
    {
        const int r = lane & 15, hh = (lane >> 4) * 16;
        for (int dc = 0; dc < 4; ++dc)
            qfrag[dc] = *(const v16bf*)&Qs[r][dc * 32 + hh];
    }

    v8f oacc[8];
    for (int i = 0; i < 8; ++i) oacc[i] = (v8f){0.f,0.f,0.f,0.f,0.f,0.f,0.f,0.f};

    const int nchunk = ((m0 + 15) >> 5) + 1;     // causal: only chunks with keys <= m0+15
    for (int kc = 0; kc < nchunk; ++kc) {
        const int ks = kc * 32;
        // stage V chunk row-major via async-to-LDS (lane = key row)
        {
            const ushort_t* vr = qkv + base + (size_t)(ks + lane) * (3 * Hc) + 2 * Hc + h * Dc;
            for (int i = 0; i < 8; ++i)
                async_copy_b128(&Vs[lane][i * 8], vr + i * 8);
        }
        // stage K chunk transposed (register path: elementwise reorder)
        {
            const ushort_t* kr = qkv + base + (size_t)(ks + lane) * (3 * Hc) + Hc + h * Dc;
            uint4 tmp[8];
            for (int i = 0; i < 8; ++i) tmp[i] = ((const uint4*)kr)[i];
            const ushort_t* tp = (const ushort_t*)tmp;
            for (int d = 0; d < Dc; ++d) Kts[d][lane] = tp[d];
        }
        smrow[lane] = smask[(size_t)b * Kc + ks + lane];
        wait_async_0();
        __syncthreads();

        // S = (Q * K^T) / sqrt(d), masked, written to LDS
        for (int nt = 0; nt < 2; ++nt) {
            v8f s = (v8f){0.f,0.f,0.f,0.f,0.f,0.f,0.f,0.f};
            for (int dc = 0; dc < 4; ++dc) {
                v16bf bfrag;
                for (int s2 = 0; s2 < 16; ++s2)
                    bfrag[s2] = __builtin_bit_cast(__bf16, Kts[dc * 32 + (lane >> 4) * 16 + s2][nt * 16 + (lane & 15)]);
                s = __builtin_amdgcn_wmma_f32_16x16x32_bf16(false, qfrag[dc], false, bfrag,
                                                            (short)0, s, false, false);
            }
            const int n_l   = nt * 16 + (lane & 15);
            const int gk    = ks + n_l;
            const int mbase = (lane >> 4) * 8;
            const bool pad  = smrow[n_l] < 0.5f;
            for (int v = 0; v < 8; ++v) {
                const int m_l = mbase + v;
                float sv = s[v] * 0.08838834764831845f;
                if (gk > m0 + m_l) sv = NEG_BIG;     // causal
                if (pad)           sv = NEG_BIG;     // slot-mask pad
                Ss[m_l][n_l] = sv;
            }
        }
        __syncthreads();

        // online softmax: lanes 0..15 each own one row
        if (lane < 16) {
            const float mo = mrow[lane];
            float cm = mo;
            for (int j = 0; j < 32; ++j) cm = fmaxf(cm, Ss[lane][j]);
            const float alpha = __expf(mo - cm);
            float lsum = 0.0f;
            for (int j = 0; j < 32; ++j) {
                const float pj = __expf(Ss[lane][j] - cm);
                Ps[lane][j] = f2bf(pj);
                lsum += pj;
            }
            lrow[lane] = lrow[lane] * alpha + lsum;
            mrow[lane] = cm;
            arow[lane] = alpha;
        }
        __syncthreads();

        // rescale accumulator, then O += P * V
        {
            const int mbase = (lane >> 4) * 8;
            for (int nt = 0; nt < 8; ++nt)
                for (int v = 0; v < 8; ++v)
                    oacc[nt][v] *= arow[mbase + v];
        }
        v16bf pfrag;
        {
            const int r = lane & 15, hh = (lane >> 4) * 16;
            pfrag = *(const v16bf*)&Ps[r][hh];
        }
        for (int nt = 0; nt < 8; ++nt) {
            v16bf vfrag;
            for (int s2 = 0; s2 < 16; ++s2)
                vfrag[s2] = __builtin_bit_cast(__bf16, Vs[(lane >> 4) * 16 + s2][nt * 16 + (lane & 15)]);
            oacc[nt] = __builtin_amdgcn_wmma_f32_16x16x32_bf16(false, pfrag, false, vfrag,
                                                               (short)0, oacc[nt], false, false);
        }
        __syncthreads();
    }

    // epilogue: divide by running denominator, store bf16
    {
        const int dn = lane & 15, mbase = (lane >> 4) * 8;
        for (int nt = 0; nt < 8; ++nt)
            for (int v = 0; v < 8; ++v) {
                const int m = mbase + v;
                const float val = oacc[nt][v] / lrow[m];
                o_out[((size_t)b * Kc + m0 + m) * Hc + h * Dc + nt * 16 + dn] = f2bf(val);
            }
    }
}

// ---------------- host orchestration ----------------
static void launch_gemm(const ushort_t* A, int lda, const ushort_t* Wt, int ldw,
                        const float* bias, float* outF, ushort_t* outB, int ldo,
                        int M, int N, int Kd, hipStream_t s) {
    dim3 grid(M / 128, N / 128), blk(256);
    gemm_bf16_kernel<<<grid, blk, 0, s>>>(A, lda, Wt, ldw, bias, outF, outB, ldo, M, N, Kd);
}

extern "C" void kernel_launch(void* const* d_in, const int* in_sizes, int n_in,
                              void* d_out, int out_size, void* d_ws, size_t ws_size,
                              hipStream_t stream) {
    (void)in_sizes; (void)n_in; (void)out_size; (void)ws_size;

    const float*     projected = (const float*)d_in[0];
    const long long* bnd       = (const long long*)d_in[1];   // int64
    const float*     smask     = (const float*)d_in[2];
    const float*     qp_w      = (const float*)d_in[3];
    const float*     qp_b      = (const float*)d_in[4];
    const float*     ca_in_w   = (const float*)d_in[5];
    const float*     ca_in_b   = (const float*)d_in[6];
    const float*     ca_out_w  = (const float*)d_in[7];
    const float*     ca_out_b  = (const float*)d_in[8];
    const float*     cn_g      = (const float*)d_in[9];
    const float*     cn_b      = (const float*)d_in[10];
    const float*     sa_in_w   = (const float*)d_in[11];
    const float*     sa_in_b   = (const float*)d_in[12];
    const float*     sa_out_w  = (const float*)d_in[13];
    const float*     sa_out_b  = (const float*)d_in[14];
    const float*     on_g      = (const float*)d_in[15];
    const float*     on_b      = (const float*)d_in[16];
    float* out = (float*)d_out;

    const size_t nProj = (size_t)Bc * Tc * Hc;          // 33.5M
    const size_t nSlot = (size_t)Bc * Kc * Hc;          // 16.8M
    const size_t nKV   = (size_t)Bc * Tc * 2 * Hc;      // 67.1M (>= B*K*3H for reuse)

    // workspace carve-out (aliased across phases)
    char* w = (char*)d_ws;
    auto alloc = [&](size_t bytes) { char* p = w; w += (bytes + 255) & ~(size_t)255; return p; };
    ushort_t* Wqp    = (ushort_t*)alloc((size_t)Hc * Hc * 2);
    ushort_t* Wcain  = (ushort_t*)alloc((size_t)Hc * 3 * Hc * 2);
    ushort_t* Wcaout = (ushort_t*)alloc((size_t)Hc * Hc * 2);
    ushort_t* Wsain  = (ushort_t*)alloc((size_t)Hc * 3 * Hc * 2);
    ushort_t* Wsaout = (ushort_t*)alloc((size_t)Hc * Hc * 2);
    ushort_t* projB  = (ushort_t*)alloc(nProj * 2);
    ushort_t* poolB  = (ushort_t*)alloc(nSlot * 2);     // init_slots, later o_ca
    float*    qF     = (float*)   alloc(nSlot * 4);     // queries fp32 (residual)
    ushort_t* qB     = (ushort_t*)alloc(nSlot * 2);     // queries bf16, later slots bf16
    ushort_t* qpB    = (ushort_t*)alloc(nSlot * 2);     // q-proj, later o_sa
    ushort_t* kvB    = (ushort_t*)alloc(nKV * 2);       // cross KV, later SA QKV
    float*    accF   = (float*)   alloc(nSlot * 4);     // attn_out, later ctx
    float*    slotsF = (float*)   alloc(nSlot * 4);

    // 1) weight conversion (pre-transposed for WMMA B-fragments)
    {
        int n1 = Hc * Hc, n3 = Hc * 3 * Hc;
        transpose_w_kernel<<<(n1 + 255) / 256, 256, 0, stream>>>(qp_w,     Wqp,    Hc,     Hc);
        transpose_w_kernel<<<(n3 + 255) / 256, 256, 0, stream>>>(ca_in_w,  Wcain,  3 * Hc, Hc);
        transpose_w_kernel<<<(n1 + 255) / 256, 256, 0, stream>>>(ca_out_w, Wcaout, Hc,     Hc);
        transpose_w_kernel<<<(n3 + 255) / 256, 256, 0, stream>>>(sa_in_w,  Wsain,  3 * Hc, Hc);
        transpose_w_kernel<<<(n1 + 255) / 256, 256, 0, stream>>>(sa_out_w, Wsaout, Hc,     Hc);
    }
    // 2) projected -> bf16
    f32_to_bf16_kernel<<<(unsigned)((nProj / 4 + 255) / 256), 256, 0, stream>>>(projected, projB, nProj / 4);
    // 3) segment-mean pooling -> init_slots (bf16)
    pool_kernel<<<Bc * Kc, 256, 0, stream>>>(projected, bnd, smask, poolB);
    // 4) queries = init_slots @ qp_w^T + qp_b          (fp32 + bf16)
    launch_gemm(poolB, Hc, Wqp, Hc, qp_b, qF, qB, Hc, Bc * Kc, Hc, Hc, stream);
    // 5) q-projection of queries (cross-attn wq, bq)   (bf16)
    launch_gemm(qB, Hc, Wcain, 3 * Hc, ca_in_b, nullptr, qpB, Hc, Bc * Kc, Hc, Hc, stream);
    // 6) K,V projections of projected                  (bf16, N=2H with column offset H)
    launch_gemm(projB, Hc, Wcain + Hc, 3 * Hc, ca_in_b + Hc, nullptr, kvB, 2 * Hc,
                Bc * Tc, 2 * Hc, Hc, stream);
    // 7) windowed cross-attention -> o_ca (reuses init_slots buffer)
    cross_attn_kernel<<<Bc * Kc, 256, 0, stream>>>(qpB, kvB, bnd, smask, poolB);
    // 8) attn_out = o_ca @ ca_out_w^T + b              (fp32)
    launch_gemm(poolB, Hc, Wcaout, Hc, ca_out_b, accF, nullptr, Hc, Bc * Kc, Hc, Hc, stream);
    // 9) slots = LN(attn_out + queries)                (fp32 + bf16)
    add_ln_kernel<<<Bc * Kc, 256, 0, stream>>>(accF, qF, cn_g, cn_b, slotsF, qB);
    // 10) SA QKV = slots @ sa_in_w^T + b               (bf16, reuses kv buffer)
    launch_gemm(qB, Hc, Wsain, 3 * Hc, sa_in_b, nullptr, kvB, 3 * Hc, Bc * Kc, 3 * Hc, Hc, stream);
    // 11) causal flash self-attention -> o_sa (reuses q-proj buffer)
    self_attn_kernel<<<Bc * NHc * (Kc / 16), 32, 0, stream>>>(kvB, smask, qpB);
    // 12) ctx = o_sa @ sa_out_w^T + b                  (fp32, reuses attn_out buffer)
    launch_gemm(qpB, Hc, Wsaout, Hc, sa_out_b, accF, nullptr, Hc, Bc * Kc, Hc, Hc, stream);
    // 13) out = LN(ctx + slots)
    add_ln_kernel<<<Bc * Kc, 256, 0, stream>>>(accF, slotsF, on_g, on_b, out, nullptr);
}
